// RepLKSSMLayer_89661737271893
// MI455X (gfx1250) — matrix operand order repaired and checked by reading, not compiled, not run on hardware
//
#include <hip/hip_runtime.h>
#include <hip/hip_bf16.h>
#include <math.h>

#define BSZ 4
#define CC  96
#define HD  64
#define WD  64
#define DIM 192
#define NST 16
#define RR  6
#define LD  4096
#define KD  4
#define SEG 64
#define NSEGS (LD / SEG)     // 64

typedef __attribute__((ext_vector_type(16))) _Float16 v16h;
typedef __attribute__((ext_vector_type(8)))  _Float16 v8h;
typedef __attribute__((ext_vector_type(8)))  float    v8f;

// ---------------------------------------------------------------- utilities
__device__ __forceinline__ int dirpos(int k, int l) {
    int ll = (k >= 2) ? (LD - 1 - l) : l;
    return (k & 1) ? (((ll & 63) << 6) | (ll >> 6)) : ll;
}

__device__ __forceinline__ float dwc_tap(const float* __restrict__ xp,
                                         const float* __restrict__ w,
                                         int k, int dil, int h, int wc) {
    int pad = dil * (k - 1) / 2;
    float s = 0.f;
    for (int i = 0; i < k; ++i) {
        int hh = h + i * dil - pad;
        if ((unsigned)hh >= HD) continue;
        const float* row = xp + hh * WD;
        for (int j = 0; j < k; ++j) {
            int ww = wc + j * dil - pad;
            if ((unsigned)ww >= WD) continue;
            s += row[ww] * w[i * k + j];
        }
    }
    return s;
}

// ------------------------------------------------- stage 0: weight convert
__global__ void cvt_weights(const float* __restrict__ ipw,
                            const float* __restrict__ opw,
                            _Float16* __restrict__ wz,
                            _Float16* __restrict__ wo) {
    int i = blockIdx.x * 256 + threadIdx.x;
    if (i < 2 * DIM * CC) wz[i] = (_Float16)ipw[i];   // (384, 96)
    if (i < CC * DIM)     wo[i] = (_Float16)opw[i];   // (96, 192)
}

// -------------------------------------- stage 1: 7 depthwise branches + BN
__global__ void dw_branches(const float* __restrict__ x,
                            const float* __restrict__ wlk, const float* __restrict__ w5,
                            const float* __restrict__ w7,  const float* __restrict__ w9,
                            const float* __restrict__ w34, const float* __restrict__ w35,
                            const float* __restrict__ w36,
                            const float* __restrict__ bnsc, const float* __restrict__ bnsh,
                            _Float16* __restrict__ yA) {
    int idx = blockIdx.x * 256 + threadIdx.x;
    if (idx >= BSZ * CC * HD * WD) return;
    int w = idx & 63;
    int h = (idx >> 6) & 63;
    int c = (idx >> 12) % CC;
    int b = idx / (CC * HD * WD);
    const float* xp = x + (size_t)(b * CC + c) * LD;

    const float* wt[7] = {wlk, w5, w7, w9, w34, w35, w36};
    const int kk[7]  = {13, 5, 7, 9, 3, 3, 3};
    const int dil[7] = { 1, 1, 2, 3, 4, 5, 6};
    float acc = 0.f;
    for (int i = 0; i < 7; ++i) {
        const float* wb = wt[i] + (size_t)c * kk[i] * kk[i];
        acc += dwc_tap(xp, wb, kk[i], dil[i], h, w) * bnsc[i * CC + c] + bnsh[i * CC + c];
    }
    yA[((size_t)(b << 12 | ((h << 6) | w))) * CC + c] = (_Float16)acc;
}

// ------------------------------------------------- WMMA GEMM (two epilogues)
// A: (M, K) f16 row-major.  Bt: (N, K) f16 row-major (= B transposed).
// MODE 0: Out[m*N + n] = acc                      (in_proj -> xz)
// MODE 1: Out[(b*CC + n)*LD + p] = clip(acc,0,6)  (out_proj -> NCHW d_out)
template <int MODE>
__global__ void wmma_gemm(const _Float16* __restrict__ A,
                          const _Float16* __restrict__ Bt,
                          float* __restrict__ Out,
                          int M, int N, int K) {
    int lane = threadIdx.x & 31;
    int wv   = threadIdx.x >> 5;
    int tile = blockIdx.x * 8 + wv;
    int tilesN = N >> 4;
    int tM = tile / tilesN;
    int tN = tile - tM * tilesN;
    if (tM * 16 >= M) return;           // uniform per wave

    int half = lane >> 4;               // 0 or 1
    int sub  = lane & 15;
    const _Float16* ap = A  + (size_t)(tM * 16 + sub) * K + half * 8;
    const _Float16* bp = Bt + (size_t)(tN * 16 + sub) * K + half * 8;

    v8f acc = {};
    for (int k0 = 0; k0 < K; k0 += 32) {
        v8h alo = *(const v8h*)(ap + k0);        // K: k0 + half*8 + [0..7]
        v8h ahi = *(const v8h*)(ap + k0 + 16);   // K: k0 + 16 + half*8 + [0..7]
        v16h av, bv;
#pragma unroll
        v8h blo = *(const v8h*)(bp + k0);        // K: k0 + half*8 + [0..7]
        v8h bhi = *(const v8h*)(bp + k0 + 16);   // K: k0 + 16 + half*8 + [0..7]
        for (int i = 0; i < 8; ++i) { av[i] = alo[i]; av[i + 8] = ahi[i]; bv[i] = blo[i]; bv[i + 8] = bhi[i]; }
        acc = __builtin_amdgcn_wmma_f32_16x16x32_f16(
            false, av, false, bv, (short)0, acc, false, false);
    }

    int col = tN * 16 + sub;
    int mb  = tM * 16 + half * 8;
    if (MODE == 0) {
#pragma unroll
        for (int r = 0; r < 8; ++r)
            Out[(size_t)(mb + r) * N + col] = acc[r];
    } else {
#pragma unroll
        for (int r = 0; r < 8; ++r) {
            int m = mb + r;
            int b = m >> 12, p = m & 4095;
            float v = acc[r];
            v = v < 0.f ? 0.f : (v > 6.f ? 6.f : v);
            Out[((size_t)(b * CC + col) << 12) + p] = v;
        }
    }
}

// ------------------------------------- stage 3: depthwise 3x3 + bias + SiLU
__global__ void dw3_silu(const float* __restrict__ xz,
                         const float* __restrict__ wc,
                         const float* __restrict__ bias,
                         float* __restrict__ xcT) {
    int idx = blockIdx.x * 256 + threadIdx.x;       // ((b*L + p)*DIM + d)
    if (idx >= BSZ * LD * DIM) return;
    int d   = idx % DIM;
    int row = idx / DIM;                            // b*L + p
    int p = row & 4095, b = row >> 12;
    int h = p >> 6, w = p & 63;
    float s = bias[d];
#pragma unroll
    for (int i = 0; i < 3; ++i) {
        int hh = h + i - 1;
        if ((unsigned)hh >= HD) continue;
#pragma unroll
        for (int j = 0; j < 3; ++j) {
            int ww = w + j - 1;
            if ((unsigned)ww >= WD) continue;
            s += xz[((size_t)(b << 12 | ((hh << 6) | ww))) * (2 * DIM) + d] *
                 wc[d * 9 + i * 3 + j];
        }
    }
    s = s / (1.f + __expf(-s));                     // SiLU
    xcT[idx] = s;                                   // (B*L, DIM) layout
}

// -------------------------------------------- stage 4: x_dbl = xs . x_proj_w
__global__ void xproj(const float* __restrict__ xcT,
                      const float* __restrict__ xpw,
                      float* __restrict__ xdbl) {
    int idx = blockIdx.x * 256 + threadIdx.x;       // ((bk*38 + c)*L + l)
    if (idx >= BSZ * KD * 38 * LD) return;
    int l = idx & 4095;
    int t = idx >> 12;                              // bk*38 + c
    int c  = t % 38;
    int bk = t / 38;
    int k = bk & 3, b = bk >> 2;
    const float* u  = xcT + ((size_t)(b << 12 | dirpos(k, l))) * DIM;
    const float* wp = xpw + (size_t)(k * 38 + c) * DIM;
    float s = 0.f;
#pragma unroll 4
    for (int d = 0; d < DIM; ++d) s += u[d] * wp[d];
    xdbl[idx] = s;
}

__global__ void zerof(float* __restrict__ p, int n) {
    int i = blockIdx.x * 256 + threadIdx.x;
    if (i < n) p[i] = 0.f;
}

// ---------------------------------------- stage 5: segmented selective scan
// Linearity: h_l = a_l h_{l-1} + b_l with a_l = exp(delta_l*A), so a segment
// composes as h_end = exp(A * sum(delta)) * h_start + h_local_end.
//
// Pass A (OUT=false): per (b,k,segment) block, scan from h=0; emit local end
//   state hseg[bk,d,s,0:16] and dsum[bk,d,s] = sum(delta).
// Pass B: serial over 64 segments, parallel over (b,k,d,n); rewrite hseg
//   in-place to hold segment ENTRY states.
// Pass C (OUT=true): re-scan each segment from its entry state, emitting
//   y_t + D*u_t accumulated (atomically, all 4 directions) into ycomb.
// Each block stages its 38x64 x_dbl slice in LDS once (no per-step barriers).
template <bool OUT>
__global__ void scan_pass(const float* __restrict__ xdbl,
                          const float* __restrict__ xcT,
                          const float* __restrict__ dtw,
                          const float* __restrict__ dtb,
                          const float* __restrict__ Alog,
                          const float* __restrict__ Dsk,
                          float* __restrict__ hseg,   // (BK, DIM, NSEGS, NST)
                          float* __restrict__ dsum,   // (BK, DIM, NSEGS)
                          float* __restrict__ ycomb) {
    __shared__ float sd[38 * SEG];                  // 9.5 KB
    int blk = blockIdx.x;                           // bk*NSEGS + s
    int s  = blk & (NSEGS - 1);
    int bk = blk >> 6;
    int k = bk & 3, b = bk >> 2;
    int d = threadIdx.x;                            // 0..191

    const float* xd = xdbl + (size_t)bk * 38 * LD + s * SEG;
    for (int i = d; i < 38 * SEG; i += DIM) {
        int c = i >> 6, l = i & 63;
        sd[i] = xd[(size_t)c * LD + l];
        if (s + 1 < NSEGS)                          // warm L2 for next segment
            __builtin_prefetch(xd + (size_t)c * LD + l + SEG, 0, 1);
    }
    __syncthreads();

    float An[NST], hn[NST];
#pragma unroll
    for (int n = 0; n < NST; ++n)
        An[n] = -__expf(Alog[((size_t)(k * DIM + d)) * NST + n]);
    size_t hsoff = (((size_t)bk * DIM + d) * NSEGS + s) * NST;
    if (OUT) {
#pragma unroll
        for (int n = 0; n < NST; ++n) hn[n] = hseg[hsoff + n];  // entry state
    } else {
#pragma unroll
        for (int n = 0; n < NST; ++n) hn[n] = 0.f;
    }
    float wr[RR];
#pragma unroll
    for (int r = 0; r < RR; ++r) wr[r] = dtw[((size_t)(k * DIM + d)) * RR + r];
    float bdt = dtb[k * DIM + d];
    float Dk  = Dsk[k * DIM + d];
    float dacc = 0.f;

    for (int l = 0; l < SEG; ++l) {
        float dr = bdt;
#pragma unroll
        for (int r = 0; r < RR; ++r) dr += sd[r * SEG + l] * wr[r];
        float delta = dr > 20.f ? dr : log1pf(__expf(dr));   // softplus
        int p = dirpos(k, s * SEG + l);
        float u  = xcT[((size_t)(b << 12 | p)) * DIM + d];
        float du = delta * u;
        if (!OUT) {
            dacc += delta;
#pragma unroll
            for (int n = 0; n < NST; ++n)
                hn[n] = __expf(delta * An[n]) * hn[n] + du * sd[(RR + n) * SEG + l];
        } else {
            float y = 0.f;
#pragma unroll
            for (int n = 0; n < NST; ++n) {
                hn[n] = __expf(delta * An[n]) * hn[n] + du * sd[(RR + n) * SEG + l];
                y += hn[n] * sd[(RR + NST + n) * SEG + l];
            }
            atomicAdd(&ycomb[((size_t)(b << 12 | p)) * DIM + d], y + Dk * u);
        }
    }
    if (!OUT) {
#pragma unroll
        for (int n = 0; n < NST; ++n) hseg[hsoff + n] = hn[n];
        dsum[((size_t)bk * DIM + d) * NSEGS + s] = dacc;
    }
}

// Pass B: serial over segments; hseg rewritten in-place to entry states.
__global__ void scan_propagate(float* __restrict__ hseg,
                               const float* __restrict__ dsum,
                               const float* __restrict__ Alog) {
    int idx = blockIdx.x * 256 + threadIdx.x;       // (bk*DIM + d)*NST + n
    if (idx >= BSZ * KD * DIM * NST) return;
    int n = idx & 15;
    int t = idx >> 4;                               // bk*DIM + d
    int d  = t % DIM;
    int bk = t / DIM;
    int k = bk & 3;
    float A = -__expf(Alog[((size_t)(k * DIM + d)) * NST + n]);
    float h = 0.f;
    size_t base = (size_t)t * NSEGS * NST + n;
    const float* ds_ = dsum + (size_t)t * NSEGS;
    for (int s = 0; s < NSEGS; ++s) {
        size_t ix = base + (size_t)s * NST;
        float hl = hseg[ix];                        // local end of segment s
        hseg[ix] = h;                               // entry state of segment s
        h = __expf(A * ds_[s]) * h + hl;
    }
}

// ------------------------------------------ stage 6: LayerNorm + SiLU(z) gate
__global__ void ln_gate(const float* __restrict__ ycomb,
                        const float* __restrict__ xz,
                        const float* __restrict__ gamma,
                        const float* __restrict__ beta,
                        _Float16* __restrict__ yo) {
    __shared__ float red[256];
    int row = blockIdx.x;                           // b*L + p
    int t = threadIdx.x;                            // 0..255
    float v = (t < DIM) ? ycomb[(size_t)row * DIM + t] : 0.f;
    red[t] = v;
    __syncthreads();
    for (int s = 128; s > 0; s >>= 1) { if (t < s) red[t] += red[t + s]; __syncthreads(); }
    float mu = red[0] / DIM;
    __syncthreads();
    float dv = (t < DIM) ? (v - mu) : 0.f;
    red[t] = dv * dv;
    __syncthreads();
    for (int s = 128; s > 0; s >>= 1) { if (t < s) red[t] += red[t + s]; __syncthreads(); }
    float var = red[0] / DIM;
    if (t < DIM) {
        float ln = dv * rsqrtf(var + 1e-5f) * gamma[t] + beta[t];
        float z  = xz[(size_t)row * (2 * DIM) + DIM + t];
        float g  = z / (1.f + __expf(-z));
        yo[(size_t)row * DIM + t] = (_Float16)(ln * g);
    }
}

// --------------------------------------------------------------------- host
extern "C" void kernel_launch(void* const* d_in, const int* in_sizes, int n_in,
                              void* d_out, int out_size, void* d_ws, size_t ws_size,
                              hipStream_t stream) {
    const float* x       = (const float*)d_in[0];
    const float* w_lk    = (const float*)d_in[1];
    const float* w_k5    = (const float*)d_in[2];
    const float* w_k7    = (const float*)d_in[3];
    const float* w_k9    = (const float*)d_in[4];
    const float* w_k3d4  = (const float*)d_in[5];
    const float* w_k3d5  = (const float*)d_in[6];
    const float* w_k3d6  = (const float*)d_in[7];
    const float* bn_sc   = (const float*)d_in[8];
    const float* bn_sh   = (const float*)d_in[9];
    const float* in_pw   = (const float*)d_in[10];
    const float* c2w     = (const float*)d_in[11];
    const float* c2b     = (const float*)d_in[12];
    const float* xpw     = (const float*)d_in[13];
    const float* dtw     = (const float*)d_in[14];
    const float* dtb     = (const float*)d_in[15];
    const float* alog    = (const float*)d_in[16];
    const float* dsk     = (const float*)d_in[17];
    const float* lng     = (const float*)d_in[18];
    const float* lnb     = (const float*)d_in[19];
    const float* out_pw  = (const float*)d_in[20];

    const int M = BSZ * LD;                         // 16384
    const int BK = BSZ * KD;                        // 16

    char* ws = (char*)d_ws;
    size_t off = 0;
    auto alloc = [&](size_t bytes) -> void* {
        void* p = ws + off;
        off = (off + bytes + 255) & ~(size_t)255;
        return p;
    };
    _Float16* yA    = (_Float16*)alloc((size_t)M * CC * 2);              // 3.1 MB
    _Float16* wz    = (_Float16*)alloc((size_t)2 * DIM * CC * 2);        // 72 KB
    _Float16* wo    = (_Float16*)alloc((size_t)CC * DIM * 2);            // 36 KB
    float*    xz    = (float*)   alloc((size_t)M * 2 * DIM * 4);         // 25 MB
    float*    xcT   = (float*)   alloc((size_t)M * DIM * 4);             // 12.6 MB
    float*    xdbl  = (float*)   alloc((size_t)BK * 38 * LD * 4);        // 10 MB
    float*    ycomb = (float*)   alloc((size_t)M * DIM * 4);             // 12.6 MB
    _Float16* yo    = (_Float16*)alloc((size_t)M * DIM * 2);             // 6.3 MB
    float*    hseg  = (float*)   alloc((size_t)BK * DIM * NSEGS * NST * 4); // 12.6 MB
    float*    dsum  = (float*)   alloc((size_t)BK * DIM * NSEGS * 4);    // 0.8 MB

    // 0) convert GEMM weights to f16 (N,K) layouts (already transposed)
    cvt_weights<<<(2 * DIM * CC + 255) / 256, 256, 0, stream>>>(in_pw, out_pw, wz, wo);

    // 1) 7 depthwise branches + BN -> f16 A-matrix (B*L, C)
    dw_branches<<<(BSZ * CC * LD + 255) / 256, 256, 0, stream>>>(
        x, w_lk, w_k5, w_k7, w_k9, w_k3d4, w_k3d5, w_k3d6, bn_sc, bn_sh, yA);

    // 2) in_proj GEMM (WMMA): (16384,96) x (96,384) -> xz
    wmma_gemm<0><<<(M / 16) * (2 * DIM / 16) / 8, 256, 0, stream>>>(
        yA, wz, xz, M, 2 * DIM, CC);

    // 3) depthwise 3x3 + bias + SiLU -> xcT (B*L, DIM)
    dw3_silu<<<(BSZ * LD * DIM + 255) / 256, 256, 0, stream>>>(xz, c2w, c2b, xcT);

    // 4) x_dbl projections for all 4 scan directions
    xproj<<<(BSZ * KD * 38 * LD + 255) / 256, 256, 0, stream>>>(xcT, xpw, xdbl);

    // 5) segmented selective scan (3 passes), all 4 directions fused
    zerof<<<(M * DIM + 255) / 256, 256, 0, stream>>>(ycomb, M * DIM);
    scan_pass<false><<<BK * NSEGS, DIM, 0, stream>>>(
        xdbl, xcT, dtw, dtb, alog, dsk, hseg, dsum, ycomb);
    scan_propagate<<<(BK * DIM * NST + 255) / 256, 256, 0, stream>>>(hseg, dsum, alog);
    scan_pass<true><<<BK * NSEGS, DIM, 0, stream>>>(
        xdbl, xcT, dtw, dtb, alog, dsk, hseg, dsum, ycomb);

    // 6) LayerNorm + SiLU(z) gate -> f16 A-matrix for out_proj
    ln_gate<<<M, 256, 0, stream>>>(ycomb, xz, lng, lnb, yo);

    // 7) out_proj GEMM (WMMA) + clip + NCHW transpose -> d_out
    wmma_gemm<1><<<(M / 16) * (CC / 16) / 8, 256, 0, stream>>>(
        yo, wo, (float*)d_out, M, CC, DIM);

    (void)in_sizes; (void)n_in; (void)out_size; (void)ws_size;
}
